// LRUBlock_49340584296999
// MI455X (gfx1250) — matrix operand order repaired and checked by reading, not compiled, not run on hardware
//
#include <hip/hip_runtime.h>
#include <stdint.h>

#define D_DIM 1024
#define B_DIM 16
#define T_DIM 2048
#define BT (B_DIM*T_DIM)            // 32768 rows
#define CHUNK 32
#define NCHUNK (T_DIM/CHUNK)        // 64 chunks
#define NLAYERS 4

typedef __attribute__((ext_vector_type(16))) __bf16 v16bf;
typedef __attribute__((ext_vector_type(8)))  float  v8f;

// ---------- helpers ----------

__device__ __forceinline__ __bf16 f2bf(float f) {
  union { float f; uint32_t u; } c; c.f = f;
  uint32_t r = c.u + 0x7FFFu + ((c.u >> 16) & 1u);   // round-to-nearest-even
  union { uint16_t s; __bf16 b; } o; o.s = (uint16_t)(r >> 16);
  return o.b;
}

union FragU { v16bf v; uint4 q[2]; };

// A-fragment (16x32 bf16, M x K): per ISA layout, lane l (m=l&15, kh=l>>4)
// holds K = {kh*8 .. kh*8+7} and {16+kh*8 .. 16+kh*8+7}  -> two 16B loads.
__device__ __forceinline__ v16bf load_a_frag(const __bf16* base, int ldk, int lane, int k0) {
  int m = lane & 15, kh = lane >> 4;
  const __bf16* p = base + m * ldk + k0 + kh * 8;
  FragU u;
  u.q[0] = *(const uint4*)(p);
  u.q[1] = *(const uint4*)(p + 16);
  return u.v;
}

// B-fragment (32x16 bf16, K x N) read from a transposed [N][K] bf16 buffer:
// lane l (n=l&15, kh=l>>4) holds K = {kh*8 .. kh*8+7} and {16+kh*8 .. 16+kh*8+7}  -> two 16B loads.
__device__ __forceinline__ v16bf load_b_frag(const __bf16* wt, int ldk, int lane, int n0, int k0) {
  int n = n0 + (lane & 15);
  int kc = k0 + ((lane >> 4) << 3);
  const __bf16* p = wt + (long)n * ldk + kc;
  FragU u;
  u.q[0] = *(const uint4*)(p);
  u.q[1] = *(const uint4*)(p + 16);
  return u.v;
}

__device__ __forceinline__ v8f wmma_bf16(v16bf a, v16bf b, v8f c) {
  return __builtin_amdgcn_wmma_f32_16x16x32_bf16(false, a, false, b, (short)0, c, false, false);
}

// gfx1250 async global->LDS copy (VGLOBAL GLOBAL_LOAD_ASYNC_TO_LDS_B128).
// vdst = LDS byte address (low 32 bits of the flat shared pointer, per the
// ISA flat->LDS aperture truncation rule), vaddr = 64-bit global address.
// Tracked by ASYNCcnt; completion enforced with s_wait_asynccnt.
__device__ __forceinline__ void async_b128(uint32_t lds_addr, const __bf16* gp) {
  asm volatile("global_load_async_to_lds_b128 %0, %1, off"
               :: "v"(lds_addr), "v"((unsigned long long)(uintptr_t)gp)
               : "memory");
}
__device__ __forceinline__ void wait_async() {
  asm volatile("s_wait_asynccnt 0x0" ::: "memory");
}

// ---------- transpose + f32->bf16 convert:  out[n*K+k] = bf16(in[k*N+n]) ----------

__global__ __launch_bounds__(256) void tconv(const float* __restrict__ in,
                                             __bf16* __restrict__ out, int K, int N) {
  __shared__ float tile[32][33];
  int bk = blockIdx.y * 32, bn = blockIdx.x * 32;
  int tx = threadIdx.x & 31, ty = threadIdx.x >> 5;       // ty: 0..7
  for (int i = 0; i < 4; ++i)
    tile[ty + i * 8][tx] = in[(long)(bk + ty + i * 8) * N + bn + tx];
  __syncthreads();
  for (int i = 0; i < 4; ++i)
    out[(long)(bn + ty + i * 8) * K + bk + tx] = f2bf(tile[tx][ty + i * 8]);
}

// ---------- GEMM: C[M,N] = A[M,K](f32) * W (bf16, stored transposed [N][K]) + bias ----------
// 256 threads = 8 waves; 128x128 tile; each wave computes 32x64 (2x4 WMMA subtiles).
// Software pipelined: B tiles stream via async global->LDS (ASYNCcnt), A tiles via a
// one-tile-ahead register pipeline; double-buffered LDS, single barrier per k-tile.

__global__ __launch_bounds__(256) void gemm_bf16(
    const float* __restrict__ A, const __bf16* __restrict__ WT,
    const float* __restrict__ bias, float* __restrict__ C,
    int M, int N, int K)
{
  __shared__ __align__(16) __bf16 sA[2][128 * 40];
  __shared__ __align__(16) __bf16 sB[2][128 * 40];
  const int tid = threadIdx.x, lane = tid & 31, wave = tid >> 5;
  const int bm = blockIdx.y * 128, bn = blockIdx.x * 128;
  const int wm = (wave & 3) * 32, wn = (wave >> 2) * 64;

  v8f acc[2][4];
  for (int i = 0; i < 2; ++i)
    for (int j = 0; j < 4; ++j)
      for (int r = 0; r < 8; ++r) acc[i][j][r] = 0.f;

  const int arow = tid >> 3, acol = (tid & 7) * 4;   // A: 32 rows/pass, 4 f32 each
  const int brow = tid >> 2, bcol = (tid & 3) * 8;   // B: 64 rows/pass, 8 bf16 each
  const int KT = K / 32;

  // prologue: kick off tile 0
  float4 fa[4];
#pragma unroll
  for (int p = 0; p < 4; ++p)
    fa[p] = *(const float4*)(A + (long)(bm + arow + p * 32) * K + acol);
#pragma unroll
  for (int p = 0; p < 2; ++p)
    async_b128((uint32_t)(uintptr_t)&sB[0][(brow + p * 64) * 40 + bcol],
               WT + (long)(bn + brow + p * 64) * K + bcol);

  for (int kt = 0; kt < KT; ++kt) {
    const int cur = kt & 1, nxt = cur ^ 1;
    // convert the prefetched A registers into LDS
#pragma unroll
    for (int p = 0; p < 4; ++p) {
      __bf16* d = &sA[cur][(arow + p * 32) * 40 + acol];
      d[0] = f2bf(fa[p].x); d[1] = f2bf(fa[p].y); d[2] = f2bf(fa[p].z); d[3] = f2bf(fa[p].w);
    }
    wait_async();                 // this wave's async B tile landed in LDS
    __syncthreads();              // tiles [cur] visible to all waves
    if (kt + 1 < KT) {            // issue next tile while computing this one
      const int k0n = (kt + 1) * 32;
#pragma unroll
      for (int p = 0; p < 2; ++p)
        async_b128((uint32_t)(uintptr_t)&sB[nxt][(brow + p * 64) * 40 + bcol],
                   WT + (long)(bn + brow + p * 64) * K + k0n + bcol);
#pragma unroll
      for (int p = 0; p < 4; ++p)
        fa[p] = *(const float4*)(A + (long)(bm + arow + p * 32) * K + k0n + acol);
    }
    v16bf af0 = load_a_frag(&sA[cur][(wm +  0) * 40], 40, lane, 0);
    v16bf af1 = load_a_frag(&sA[cur][(wm + 16) * 40], 40, lane, 0);
#pragma unroll
    for (int j = 0; j < 4; ++j) {
      v16bf bf = load_b_frag(&sB[cur][0], 40, lane, wn + j * 16, 0);
      acc[0][j] = wmma_bf16(af0, bf, acc[0][j]);
      acc[1][j] = wmma_bf16(af1, bf, acc[1][j]);
    }
  }

  const int nl = lane & 15, kh = lane >> 4;
  for (int i = 0; i < 2; ++i)
    for (int j = 0; j < 4; ++j) {
      int gn = bn + wn + j * 16 + nl;
      float bv = bias ? bias[gn] : 0.f;
#pragma unroll
      for (int r = 0; r < 8; ++r) {
        int gm = bm + wm + i * 16 + r + 8 * kh;
        C[(long)gm * N + gn] = acc[i][j][r] + bv;
      }
    }
}

// ---------- Phase 2a: chunk-local scan. One WG per chunk; h kept in LDS (bf16), ----------
// ---------- accumulation fp32. y <- local prefix; carryEnd <- chunk-final state. ----------

__global__ __launch_bounds__(256) void local_scan(
    const float* __restrict__ xp, const __bf16* __restrict__ WhT,
    float* __restrict__ y, float* __restrict__ carryEnd)
{
  __shared__ __align__(16) __bf16 hb[2][B_DIM * D_DIM];   // 64 KB double buffer
  const int c = blockIdx.x;
  const int tid = threadIdx.x, lane = tid & 31, wave = tid >> 5;
  const int nl = lane & 15, kh = lane >> 4;

  for (int j = 0; j < CHUNK; ++j) {
    const int t = c * CHUNK + j;
    const int cur = j & 1, prv = cur ^ 1;
    v8f acc[8];
    for (int nt = 0; nt < 8; ++nt)
      for (int r = 0; r < 8; ++r) acc[nt][r] = 0.f;
    if (j > 0) {
      for (int k0 = 0; k0 < D_DIM; k0 += 32) {
        v16bf a = load_a_frag(&hb[prv][0], D_DIM, lane, k0);
#pragma unroll
        for (int nt = 0; nt < 8; ++nt)
          acc[nt] = wmma_bf16(a, load_b_frag(WhT, D_DIM, lane, wave * 128 + nt * 16, k0), acc[nt]);
      }
    }
    for (int nt = 0; nt < 8; ++nt) {
      const int gn = wave * 128 + nt * 16 + nl;
#pragma unroll
      for (int r = 0; r < 8; ++r) {
        const int b = r + 8 * kh;                       // batch row == WMMA M
        const long idx = ((long)b * T_DIM + t) * D_DIM + gn;
        float v = acc[nt][r] + xp[idx];
        y[idx] = v;
        hb[cur][b * D_DIM + gn] = f2bf(v);
        if (j == CHUNK - 1) carryEnd[((long)c * B_DIM + b) * D_DIM + gn] = v;
      }
    }
    __syncthreads();
  }
}

// ---------- Phase 2b: serial carry chain over chunks: H_{c} = carryEnd_{c-1} + H_{c-1} @ Wh^CHUNK ----------

__global__ __launch_bounds__(256) void carry_chain(
    const float* __restrict__ carryEnd, const __bf16* __restrict__ WhCT,
    float* __restrict__ Hbuf)
{
  __shared__ __align__(16) __bf16 hb[B_DIM * D_DIM];
  const int tid = threadIdx.x, lane = tid & 31, wave = tid >> 5;
  const int nl = lane & 15, kh = lane >> 4;
  for (int i = tid; i < B_DIM * D_DIM; i += 256) { hb[i] = f2bf(0.f); Hbuf[i] = 0.f; }
  __syncthreads();
  for (int c = 1; c < NCHUNK; ++c) {
    v8f acc[8];
    for (int nt = 0; nt < 8; ++nt)
      for (int r = 0; r < 8; ++r) acc[nt][r] = 0.f;
    for (int k0 = 0; k0 < D_DIM; k0 += 32) {
      v16bf a = load_a_frag(hb, D_DIM, lane, k0);
#pragma unroll
      for (int nt = 0; nt < 8; ++nt)
        acc[nt] = wmma_bf16(a, load_b_frag(WhCT, D_DIM, lane, wave * 128 + nt * 16, k0), acc[nt]);
    }
    __syncthreads();   // everyone done reading previous H before overwrite
    for (int nt = 0; nt < 8; ++nt) {
      const int gn = wave * 128 + nt * 16 + nl;
#pragma unroll
      for (int r = 0; r < 8; ++r) {
        const int b = r + 8 * kh;
        float v = acc[nt][r] + carryEnd[((long)(c - 1) * B_DIM + b) * D_DIM + gn];
        Hbuf[((long)c * B_DIM + b) * D_DIM + gn] = v;
        hb[b * D_DIM + gn] = f2bf(v);
      }
    }
    __syncthreads();
  }
}

// ---------- Phase 2c: propagate carries into each chunk: y_{cC+j} += H_c @ Wh^{j+1} ----------

__global__ __launch_bounds__(256) void propagate(
    const float* __restrict__ Hbuf, const __bf16* __restrict__ WhT,
    float* __restrict__ y)
{
  const int c = blockIdx.x;
  if (c == 0) return;                                   // H_0 == 0
  __shared__ __align__(16) __bf16 ub[2][B_DIM * D_DIM];
  const int tid = threadIdx.x, lane = tid & 31, wave = tid >> 5;
  const int nl = lane & 15, kh = lane >> 4;
  for (int i = tid; i < B_DIM * D_DIM; i += 256)
    ub[0][i] = f2bf(Hbuf[(long)c * B_DIM * D_DIM + i]);
  __syncthreads();
  for (int j = 0; j < CHUNK; ++j) {
    const int t = c * CHUNK + j;
    const int cur = j & 1, nxt = cur ^ 1;
    v8f acc[8];
    for (int nt = 0; nt < 8; ++nt)
      for (int r = 0; r < 8; ++r) acc[nt][r] = 0.f;
    for (int k0 = 0; k0 < D_DIM; k0 += 32) {
      v16bf a = load_a_frag(&ub[cur][0], D_DIM, lane, k0);
#pragma unroll
      for (int nt = 0; nt < 8; ++nt)
        acc[nt] = wmma_bf16(a, load_b_frag(WhT, D_DIM, lane, wave * 128 + nt * 16, k0), acc[nt]);
    }
    for (int nt = 0; nt < 8; ++nt) {
      const int gn = wave * 128 + nt * 16 + nl;
#pragma unroll
      for (int r = 0; r < 8; ++r) {
        const int b = r + 8 * kh;
        const long idx = ((long)b * T_DIM + t) * D_DIM + gn;
        float v = acc[nt][r];
        y[idx] += v;
        ub[nxt][b * D_DIM + gn] = f2bf(v);
      }
    }
    __syncthreads();
  }
}

// ---------- BatchNorm: deterministic two-stage reduction + apply(gelu)+residual ----------

__global__ __launch_bounds__(256) void bn_partial(
    const float* __restrict__ z, float* __restrict__ part, int rowsPerBlk)
{
  const int tid = threadIdx.x;
  const long r0 = (long)blockIdx.x * rowsPerBlk;
  float s1[4] = {0.f, 0.f, 0.f, 0.f}, s2[4] = {0.f, 0.f, 0.f, 0.f};
  for (int r = 0; r < rowsPerBlk; ++r) {
    const float* row = z + (r0 + r) * D_DIM;
#pragma unroll
    for (int q = 0; q < 4; ++q) {
      float v = row[tid + q * 256];
      s1[q] += v; s2[q] += v * v;
    }
  }
  for (int q = 0; q < 4; ++q) {
    int d = tid + q * 256;
    part[(long)blockIdx.x * 2 * D_DIM + d]          = s1[q];
    part[(long)blockIdx.x * 2 * D_DIM + D_DIM + d]  = s2[q];
  }
}

__global__ __launch_bounds__(256) void bn_reduce(
    const float* __restrict__ part, float* __restrict__ stats, int nblk)
{
  int e = blockIdx.x * 256 + threadIdx.x;   // 0..2*D-1 (sum then sumsq)
  float s = 0.f;
  for (int b = 0; b < nblk; ++b) s += part[(long)b * 2 * D_DIM + e];
  stats[e] = s;
}

__global__ __launch_bounds__(256) void bn_apply(
    const float* __restrict__ z, const float* __restrict__ stats,
    const float* __restrict__ gamma, const float* __restrict__ beta,
    float* __restrict__ y)
{
  const long i = (long)blockIdx.x * 256 + threadIdx.x;
  const int d = (int)(i & (D_DIM - 1));
  const float invN = 1.f / (float)BT;
  float mean = stats[d] * invN;
  float var  = stats[D_DIM + d] * invN - mean * mean;
  float zn = (z[i] - mean) * rsqrtf(var + 1e-5f) * gamma[d] + beta[d];
  float g = 0.5f * zn * (1.f + tanhf(0.7978845608028654f * (zn + 0.044715f * zn * zn * zn)));
  y[i] += g;
}

// ---------- host orchestration ----------

extern "C" void kernel_launch(void* const* d_in, const int* in_sizes, int n_in,
                              void* d_out, int out_size, void* d_ws, size_t ws_size,
                              hipStream_t stream)
{
  (void)in_sizes; (void)n_in; (void)out_size; (void)ws_size;
  const float* x    = (const float*)d_in[0];
  const float* Wi   = (const float*)d_in[1];
  const float* bi   = (const float*)d_in[2];
  const float* Wh   = (const float*)d_in[3];
  const float* mlpW = (const float*)d_in[4];
  const float* mlpb = (const float*)d_in[5];
  const float* bns  = (const float*)d_in[6];
  const float* bnb  = (const float*)d_in[7];
  float* y = (float*)d_out;

  uint8_t* ws = (uint8_t*)d_ws;
  size_t off = 0;
  auto carve = [&](size_t bytes) -> void* {
    void* p = ws + off;
    off += (bytes + 255) & ~(size_t)255;
    return p;
  };
  float*  xp       = (float*) carve((size_t)BT * D_DIM * 4);
  float*  zbuf     = (float*) carve((size_t)BT * D_DIM * 4);
  float*  carryEnd = (float*) carve((size_t)NCHUNK * B_DIM * D_DIM * 4);
  float*  Hbuf     = (float*) carve((size_t)NCHUNK * B_DIM * D_DIM * 4);
  __bf16* WiT      = (__bf16*)carve((size_t)D_DIM * D_DIM * 2);
  __bf16* WhT      = (__bf16*)carve((size_t)D_DIM * D_DIM * 2);
  __bf16* WhCT     = (__bf16*)carve((size_t)D_DIM * D_DIM * 2);
  __bf16* PT       = (__bf16*)carve((size_t)D_DIM * D_DIM * 2);
  __bf16* mlpWT    = (__bf16*)carve((size_t)NLAYERS * D_DIM * D_DIM * 2);
  float*  P0       = (float*) carve((size_t)D_DIM * D_DIM * 4);
  float*  P1       = (float*) carve((size_t)D_DIM * D_DIM * 4);
  float*  part     = (float*) carve((size_t)256 * 2 * D_DIM * 4);
  float*  stats    = (float*) carve((size_t)2 * D_DIM * 4);

  dim3 blk(256);
  dim3 tcg(D_DIM / 32, D_DIM / 32);

  // bf16-transposed weights (B operands)
  tconv<<<tcg, blk, 0, stream>>>(Wi, WiT, D_DIM, D_DIM);
  tconv<<<tcg, blk, 0, stream>>>(Wh, WhT, D_DIM, D_DIM);
  for (int l = 0; l < NLAYERS; ++l)
    tconv<<<tcg, blk, 0, stream>>>(mlpW + (size_t)l * D_DIM * D_DIM,
                                   mlpWT + (size_t)l * D_DIM * D_DIM, D_DIM, D_DIM);

  // xp = x @ Wi + bi
  gemm_bf16<<<dim3(D_DIM / 128, BT / 128), blk, 0, stream>>>(x, WiT, bi, xp, BT, D_DIM, D_DIM);

  // Wh^CHUNK by repeated squaring (CHUNK=32 -> 5 squarings)
  const float*  curA = Wh;
  const __bf16* curT = WhT;
  for (int it = 0; it < 5; ++it) {
    float* nxt = (it & 1) ? P1 : P0;
    gemm_bf16<<<dim3(D_DIM / 128, D_DIM / 128), blk, 0, stream>>>(curA, curT, nullptr, nxt,
                                                                  D_DIM, D_DIM, D_DIM);
    __bf16* nt = (it == 4) ? WhCT : PT;
    tconv<<<tcg, blk, 0, stream>>>(nxt, nt, D_DIM, D_DIM);
    curA = nxt; curT = nt;
  }

  // chunked parallel scan for the linear recurrence
  local_scan<<<NCHUNK, blk, 0, stream>>>(xp, WhT, y, carryEnd);
  carry_chain<<<1, blk, 0, stream>>>(carryEnd, WhCT, Hbuf);
  propagate<<<NCHUNK, blk, 0, stream>>>(Hbuf, WhT, y);

  // residual MLP stack
  for (int l = 0; l < NLAYERS; ++l) {
    gemm_bf16<<<dim3(D_DIM / 128, BT / 128), blk, 0, stream>>>(
        y, mlpWT + (size_t)l * D_DIM * D_DIM, mlpb + l * D_DIM, zbuf, BT, D_DIM, D_DIM);
    bn_partial<<<256, blk, 0, stream>>>(zbuf, part, BT / 256);
    bn_reduce<<<(2 * D_DIM) / 256, blk, 0, stream>>>(part, stats, 256);
    bn_apply<<<(BT * D_DIM) / 256, blk, 0, stream>>>(zbuf, stats, bns + l * D_DIM,
                                                     bnb + l * D_DIM, y);
  }
}
